// EMAQuantiser_78417512890961
// MI455X (gfx1250) — matrix operand order repaired and checked by reading, not compiled, not run on hardware
//
#include <hip/hip_runtime.h>
#include <hip/hip_bf16.h>

#define DECAY   0.99f
#define KCODES  8192
#define DIM     512
#define NROWS   32768           // 8 * 4096
#define MBLK    128             // z rows per block
#define NT      64              // codebook rows per chunk
#define NCHUNK  (KCODES / NT)   // 128
#define ZPITCH  528             // bf16 elems per LDS row (512 + 16 pad: breaks bank conflicts)

typedef __attribute__((ext_vector_type(16))) __bf16 v16bf;
typedef __attribute__((ext_vector_type(8)))  float  v8f;
typedef __attribute__((ext_vector_type(4)))  unsigned int v4u;
typedef __attribute__((ext_vector_type(8)))  int    v8i;
typedef __attribute__((ext_vector_type(4)))  int    v4i;

__device__ __forceinline__ unsigned short f2bf(float f) {
  unsigned int u = __float_as_uint(f);
  u += 0x7FFFu + ((u >> 16) & 1u);        // round-to-nearest-even
  return (unsigned short)(u >> 16);
}

__device__ __forceinline__ float waveSum(float v) {
#pragma unroll
  for (int m = 16; m >= 1; m >>= 1) v += __shfl_xor(v, m, 32);
  return v;
}

// TDM: DMA a [NT][DIM] bf16 tile (row-major, stride DIM) from global into LDS,
// with hardware row padding: 8 DWORDs (32 B = 16 bf16) after every 256 DWORDs
// (1024 B = one 512-elem bf16 row)  ->  LDS row pitch = ZPITCH = 528 bf16.
// This toolchain exposes the 6-operand builtin:
//   (v4u group0, v8i group1, v4i group2, v4i group3, v8i group4, i32 cpol)
__device__ __forceinline__ void tdm_load_chunk(const void* gsrc, unsigned lds_off) {
  const unsigned long long ga = (unsigned long long)(size_t)gsrc;
  v4u g0;
  g0.x = 1u;                                                  // count=1, user descriptor
  g0.y = lds_off;                                             // lds_addr (bytes)
  g0.z = (unsigned)(ga & 0xFFFFFFFFu);                        // global_addr[31:0]
  g0.w = (unsigned)((ga >> 32) & 0x1FFFFFFu) | (2u << 30);    // global_addr[56:32] | type=2
  v8i g1;
  g1[0] = (1 << 16)          // data_size = 2 bytes (bf16)
        | (1 << 20)          // pad_enable
        | (7 << 22)          // pad_interval code 7 = 256 DWORDs
        | (7 << 25);         // pad_amount  code 7 = 8 DWORDs
  g1[1] = (DIM & 0xFFFF) << 16;      // tensor_dim0[15:0]  @ bits 63:48
  g1[2] = (KCODES & 0xFFFF) << 16;   // tensor_dim1[15:0]  @ bits 95:80
  g1[3] = (DIM & 0xFFFF) << 16;      // tile_dim0          @ bits 127:112
  g1[4] = NT;                        // tile_dim1          @ bits 143:128
  g1[5] = DIM;                       // tensor_dim0_stride[31:0]
  g1[6] = 0;
  g1[7] = 0;
  v4i g2 = {0, 0, 0, 0};
  v4i g3 = {0, 0, 0, 0};
  v8i g4 = {0, 0, 0, 0, 0, 0, 0, 0};
  __builtin_amdgcn_tensor_load_to_lds(g0, g1, g2, g3, g4, 0);
}

// ---------------- codebook f32 -> bf16 (packed) + ||c||^2 -------------------
__global__ void k_prep_codebook(const float* __restrict__ cb,
                                unsigned int* __restrict__ cbw,   // [K][DIM/2] packed bf16 pairs
                                float* __restrict__ c2) {
  const int k = blockIdx.x, tid = threadIdx.x;   // 256 threads, one float2 each
  const float2 v = ((const float2*)(cb + (size_t)k * DIM))[tid];
  cbw[(size_t)k * (DIM / 2) + tid] =
      (unsigned int)f2bf(v.x) | ((unsigned int)f2bf(v.y) << 16);
  float s = waveSum(v.x * v.x + v.y * v.y);
  __shared__ float red[8];
  if ((tid & 31) == 0) red[tid >> 5] = s;
  __syncthreads();
  if (tid == 0) {
    float t = 0.f;
#pragma unroll
    for (int i = 0; i < 8; ++i) t += red[i];
    c2[k] = t;
  }
}

// ---------------- ||z||^2 per row ------------------------------------------
__global__ void k_row_sq(const float* __restrict__ z, float* __restrict__ z2) {
  const int r = blockIdx.x, tid = threadIdx.x;   // 128 threads, one float4 each
  const float4 v = ((const float4*)(z + (size_t)r * DIM))[tid];
  float s = waveSum(v.x * v.x + v.y * v.y + v.z * v.z + v.w * v.w);
  __shared__ float red[4];
  if ((tid & 31) == 0) red[tid >> 5] = s;
  __syncthreads();
  if (tid == 0) z2[r] = red[0] + red[1] + red[2] + red[3];
}

// ---------------- fused distance GEMM (bf16 WMMA) + argmin ------------------
__global__ void __launch_bounds__(256, 1)
k_argmin(const float* __restrict__ z, const unsigned int* __restrict__ cbw,
         const float* __restrict__ c2, const float* __restrict__ z2,
         int* __restrict__ idx_out, float* __restrict__ idxf_out) {
  extern __shared__ char smem[];
  unsigned short* zt   = (unsigned short*)smem;                                 // [MBLK][ZPITCH]
  unsigned short* cbt0 = (unsigned short*)(smem + (size_t)MBLK * ZPITCH * 2);   // [NT][ZPITCH]
  unsigned short* cbt1 = cbt0 + (size_t)NT * ZPITCH;                            // [NT][ZPITCH]

  const int tid  = threadIdx.x;
  const int wave = tid >> 5;
  const int lane = tid & 31;
  const int half = lane >> 4;     // wave32 split: lanes 0-15 / 16-31
  const int ln   = lane & 15;
  const int row0 = blockIdx.x * MBLK;
  const int rowbase = row0 + wave * 16;

  const unsigned cbt_off0 = (unsigned)(size_t)(void*)cbt0;   // LDS byte offsets for TDM
  const unsigned cbt_off1 = (unsigned)(size_t)(void*)cbt1;
  const char* cbg_bytes = (const char*)cbw;

  // Kick off TDM DMA of codebook chunk 0 while we stage the z tile.
  if (wave == 0) tdm_load_chunk(cbg_bytes, cbt_off0);

  // Stage 128x512 z tile into LDS as bf16 (loaded from HBM exactly once).
  const float4* zg = (const float4*)(z + (size_t)row0 * DIM);
#pragma unroll 4
  for (int i = tid; i < MBLK * DIM / 4; i += 256) {
    const int r = i >> 7, c4 = i & 127;
    const float4 v = zg[r * (DIM / 4) + c4];
    const unsigned int p0 = (unsigned int)f2bf(v.x) | ((unsigned int)f2bf(v.y) << 16);
    const unsigned int p1 = (unsigned int)f2bf(v.z) | ((unsigned int)f2bf(v.w) << 16);
    *(uint2*)&zt[r * ZPITCH + c4 * 4] = make_uint2(p0, p1);
  }

  // Per-lane fp32 row norms (C/D layout: VGPR r -> M = r + half*8).
  float z2v[8];
#pragma unroll
  for (int r = 0; r < 8; ++r) z2v[r] = z2[rowbase + r + half * 8];

  float bestv[8];
  int   besti[8];
#pragma unroll
  for (int r = 0; r < 8; ++r) { bestv[r] = 3.4e38f; besti[r] = 0; }

  const int kbA = half * 16;                     // 16-bit A layout K offset (contiguous, matches B)
  const int kbB = half * 16;                     // 16-bit B layout K offset
  const unsigned short* zrow = &zt[(wave * 16 + ln) * ZPITCH];

  for (int c = 0; c < NCHUNK; ++c) {
    const int c0 = c * NT;
    __syncthreads();   // all waves done reading the buffer the next DMA will overwrite
    if (wave == 0) {
      if (c + 1 < NCHUNK) {
        tdm_load_chunk(cbg_bytes + (size_t)(c0 + NT) * DIM * 2,
                       (c & 1) ? cbt_off0 : cbt_off1);
        __builtin_amdgcn_s_wait_tensorcnt(1);    // chunk c landed (TDM in-order per wave)
      } else {
        __builtin_amdgcn_s_wait_tensorcnt(0);
      }
    }
    __syncthreads();   // publish TDM-written LDS to all waves
    const unsigned short* cbuf = (c & 1) ? cbt1 : cbt0;

    v8f acc[4] = {};
#pragma unroll 2
    for (int kc = 0; kc < DIM; kc += 32) {
      // Load A fragment + all four B fragments, then fire 4 WMMAs back-to-back
      // (one s_wait_dscnt per k-step instead of four).
      union { uint4 u[2]; v16bf v; } ua, ub0, ub1, ub2, ub3;
      ua.u[0] = *(const uint4*)&zrow[kc + kbA];
      ua.u[1] = *(const uint4*)&zrow[kc + kbA + 8];
      const unsigned short* cr = &cbuf[ln * ZPITCH + kc + kbB];
      ub0.u[0] = *(const uint4*)&cr[0 * ZPITCH];
      ub0.u[1] = *(const uint4*)&cr[0 * ZPITCH + 8];
      ub1.u[0] = *(const uint4*)&cr[16 * ZPITCH];
      ub1.u[1] = *(const uint4*)&cr[16 * ZPITCH + 8];
      ub2.u[0] = *(const uint4*)&cr[32 * ZPITCH];
      ub2.u[1] = *(const uint4*)&cr[32 * ZPITCH + 8];
      ub3.u[0] = *(const uint4*)&cr[48 * ZPITCH];
      ub3.u[1] = *(const uint4*)&cr[48 * ZPITCH + 8];
      acc[0] = __builtin_amdgcn_wmma_f32_16x16x32_bf16(false, ua.v, false, ub0.v,
                                                       (short)0, acc[0], false, false);
      acc[1] = __builtin_amdgcn_wmma_f32_16x16x32_bf16(false, ua.v, false, ub1.v,
                                                       (short)0, acc[1], false, false);
      acc[2] = __builtin_amdgcn_wmma_f32_16x16x32_bf16(false, ua.v, false, ub2.v,
                                                       (short)0, acc[2], false, false);
      acc[3] = __builtin_amdgcn_wmma_f32_16x16x32_bf16(false, ua.v, false, ub3.v,
                                                       (short)0, acc[3], false, false);
    }

    // dist = ||z||^2 + ||c||^2 - 2*logit; lane-local running argmin.
#pragma unroll
    for (int t = 0; t < 4; ++t) {
      const int code = c0 + t * 16 + ln;
      const float cc = c2[code];               // tiny, L2-hot
#pragma unroll
      for (int r = 0; r < 8; ++r) {
        const float d = z2v[r] + cc - 2.0f * acc[t][r];
        if (d < bestv[r]) { bestv[r] = d; besti[r] = code; }
      }
    }
  }

  // Cross-lane min within each 16-lane half (xor 1/2/4/8 stays in-half).
#pragma unroll
  for (int r = 0; r < 8; ++r) {
    float v = bestv[r];
    int   i = besti[r];
#pragma unroll
    for (int m = 8; m >= 1; m >>= 1) {
      const float ov = __shfl_xor(v, m, 32);
      const int   oi = __shfl_xor(i, m, 32);
      if (ov < v || (ov == v && oi < i)) { v = ov; i = oi; }
    }
    if (ln == 0) {
      const int row = rowbase + r + half * 8;
      idx_out[row]  = i;
      idxf_out[row] = (float)i;
    }
  }
}

// ---------------- gather z_q, scatter segment sums, loss --------------------
__global__ void k_gather_scatter(const float* __restrict__ z, const float* __restrict__ cb,
                                 const int* __restrict__ idx,
                                 float* __restrict__ zq, float* __restrict__ sums,
                                 float* __restrict__ counts, float* __restrict__ lossw) {
  const int row = blockIdx.x, tid = threadIdx.x;   // 128 threads, one float4 each
  const int k = idx[row];
  const float4 zv = ((const float4*)(z  + (size_t)row * DIM))[tid];
  const float4 cv = ((const float4*)(cb + (size_t)k   * DIM))[tid];
  ((float4*)(zq + (size_t)row * DIM))[tid] = cv;   // z_q_st == z_q numerically
  float* sp = sums + (size_t)k * DIM + tid * 4;
  atomicAdd(sp + 0, zv.x); atomicAdd(sp + 1, zv.y);
  atomicAdd(sp + 2, zv.z); atomicAdd(sp + 3, zv.w);
  const float dx = cv.x - zv.x, dy = cv.y - zv.y, dz = cv.z - zv.z, dw = cv.w - zv.w;
  float s = waveSum(dx * dx + dy * dy + dz * dz + dw * dw);
  __shared__ float red[4];
  if ((tid & 31) == 0) red[tid >> 5] = s;
  __syncthreads();
  if (tid == 0) {
    atomicAdd(lossw, red[0] + red[1] + red[2] + red[3]);
    atomicAdd(counts + k, 1.0f);
  }
}

// ---------------- EMA finalize ---------------------------------------------
__global__ void k_finalize(const float* __restrict__ ema_count, const float* __restrict__ ema_sum,
                           const float* __restrict__ counts, const float* __restrict__ sums,
                           float* __restrict__ out_cb, float* __restrict__ out_cnt,
                           float* __restrict__ out_sum, float* __restrict__ out_usage) {
  const int k = blockIdx.x, tid = threadIdx.x;     // 128 threads, one float4 each
  const float cnt = counts[k];
  const float nc = ema_count[k] * DECAY + cnt * (1.0f - DECAY);
  if (tid == 0) {
    out_cnt[k]   = nc;
    out_usage[k] = cnt / (float)NROWS;             // total counts == NROWS by construction
  }
  const float inv = 1.0f / fmaxf(nc, 1.0f);
  const float4 es = ((const float4*)(ema_sum + (size_t)k * DIM))[tid];
  const float4 sv = ((const float4*)(sums    + (size_t)k * DIM))[tid];
  float4 ns;
  ns.x = es.x * DECAY + sv.x * (1.0f - DECAY);
  ns.y = es.y * DECAY + sv.y * (1.0f - DECAY);
  ns.z = es.z * DECAY + sv.z * (1.0f - DECAY);
  ns.w = es.w * DECAY + sv.w * (1.0f - DECAY);
  ((float4*)(out_sum + (size_t)k * DIM))[tid] = ns;
  const float4 nb = { ns.x * inv, ns.y * inv, ns.z * inv, ns.w * inv };
  ((float4*)(out_cb + (size_t)k * DIM))[tid] = nb;
}

// ---------------- scalar losses --------------------------------------------
__global__ void k_scalars(const float* __restrict__ lossw,
                          float* __restrict__ out_lc, float* __restrict__ out_le) {
  const float l = lossw[0] / (float)((size_t)NROWS * DIM);
  *out_lc = l;   // commit and embed losses are numerically identical
  *out_le = l;
}

extern "C" void kernel_launch(void* const* d_in, const int* in_sizes, int n_in,
                              void* d_out, int out_size, void* d_ws, size_t ws_size,
                              hipStream_t stream) {
  const float* z         = (const float*)d_in[0];
  const float* cb        = (const float*)d_in[1];
  const float* ema_count = (const float*)d_in[2];
  const float* ema_sum   = (const float*)d_in[3];
  float* out = (float*)d_out;

  // Output layout: z_q_st, indices, loss_commit, loss_embed, usage,
  //                new_codebook, new_ema_count, new_ema_sum
  const size_t o_zq    = 0;
  const size_t o_idx   = o_zq + (size_t)NROWS * DIM;
  const size_t o_lc    = o_idx + NROWS;
  const size_t o_le    = o_lc + 1;
  const size_t o_usage = o_le + 1;
  const size_t o_ncb   = o_usage + KCODES;
  const size_t o_ncnt  = o_ncb + (size_t)KCODES * DIM;
  const size_t o_nsum  = o_ncnt + KCODES;

  // Workspace carve-up (~25.5 MB)
  char* ws = (char*)d_ws;
  size_t off = 0;
  auto take = [&](size_t bytes) {
    char* p = ws + off;
    off += (bytes + 255) & ~(size_t)255;
    return p;
  };
  unsigned int* cbw    = (unsigned int*)take((size_t)KCODES * DIM * 2);
  float*        c2     = (float*)take((size_t)KCODES * 4);
  float*        z2     = (float*)take((size_t)NROWS * 4);
  int*          idx    = (int*)take((size_t)NROWS * 4);
  float*        counts = (float*)take((size_t)KCODES * 4);
  float*        sums   = (float*)take((size_t)KCODES * DIM * 4);
  float*        lossw  = (float*)take(4);

  (void)hipMemsetAsync(counts, 0, (size_t)KCODES * 4, stream);
  (void)hipMemsetAsync(sums,   0, (size_t)KCODES * DIM * 4, stream);
  (void)hipMemsetAsync(lossw,  0, 4, stream);

  k_prep_codebook<<<KCODES, 256, 0, stream>>>(cb, cbw, c2);
  k_row_sq<<<NROWS, 128, 0, stream>>>(z, z2);

  const size_t shmem = (size_t)(MBLK + 2 * NT) * ZPITCH * 2;   // ~270 KB of 320 KB
  k_argmin<<<NROWS / MBLK, 256, shmem, stream>>>(z, cbw, c2, z2, idx, out + o_idx);

  k_gather_scatter<<<NROWS, 128, 0, stream>>>(z, cb, idx, out + o_zq, sums, counts, lossw);
  k_finalize<<<KCODES, 128, 0, stream>>>(ema_count, ema_sum, counts, sums,
                                         out + o_ncb, out + o_ncnt, out + o_nsum,
                                         out + o_usage);
  k_scalars<<<1, 1, 0, stream>>>(lossw, out + o_lc, out + o_le);
}
